// SpatialTransformerSimpleV2_74955769249868
// MI455X (gfx1250) — matrix operand order repaired and checked by reading, not compiled, not run on hardware
//
#include <hip/hip_runtime.h>
#include <hip/hip_bf16.h>
#include <math.h>

// ---------------------------------------------------------------------------
// CDNA5 WMMA types / helpers
// ---------------------------------------------------------------------------
typedef __attribute__((ext_vector_type(16))) _Float16 v16h;
typedef __attribute__((ext_vector_type(8)))  float    v8f;

union AF {
  v16h v;
  unsigned int u[8];
  _Float16 h[16];
};

static __device__ __forceinline__ v8f wmma16(AF a, AF b, v8f c) {
  // D = A(16x32 f16) * B(32x16 f16) + C(16x16 f32)
  return __builtin_amdgcn_wmma_f32_16x16x32_f16(false, a.v, false, b.v,
                                                (short)0, c, false, false);
}

// Fragment index math (from CDNA5 ISA, 16-bit layouts, wave32):
//  A 16x32:  m = lane&15, element-pair j (VGPR j): k = 2j + (j>=4?8:0) + (lane>=16?8:0)
//  B 32x16:  n = lane&15, element-pair j:          k as in A (V^T/P^T in k_attention: 2j + (lane>=16?16:0))
//  C 16x16 f32: reg r: row = r + (lane>=16?8:0), col = lane&15

// ---------------------------------------------------------------------------
// Weight f32 -> f16 conversion (optionally transposed). dst is K x N row-major.
// ---------------------------------------------------------------------------
__global__ void k_convert_w(const float* __restrict__ src, _Float16* __restrict__ dst,
                            int K, int N, int transpose) {
  int idx = blockIdx.x * blockDim.x + threadIdx.x;
  if (idx >= K * N) return;
  int k = idx / N, n = idx % N;
  float v = transpose ? src[(size_t)n * K + k] : src[idx];
  dst[idx] = (_Float16)v;
}

// ---------------------------------------------------------------------------
// s[b,c] = 1 + sum_g cond[b,g] * W[g,c]
// ---------------------------------------------------------------------------
__global__ void k_vecmat_bias1(const float* __restrict__ cond, const float* __restrict__ W,
                               float* __restrict__ outp, int Bn, int G, int C) {
  int idx = blockIdx.x * blockDim.x + threadIdx.x;
  if (idx >= Bn * C) return;
  int b = idx / C, c = idx % C;
  float acc = 1.0f;
  for (int g = 0; g < G; ++g) acc += cond[(size_t)b * G + g] * W[(size_t)g * C + c];
  outp[idx] = acc;
}

// ---------------------------------------------------------------------------
// (b,C,HW) f32 -> (b*HW, C) f16 tokens, optional RMS norm with per-(b,c) scale.
// one block (512 thr) per token.
// ---------------------------------------------------------------------------
__global__ void __launch_bounds__(512)
k_to_tokens_f16(const float* __restrict__ x, const float* __restrict__ scale,
                _Float16* __restrict__ dst, int C, int HW, int do_norm) {
  __shared__ float red[512];
  int tid = threadIdx.x;
  int row = blockIdx.x;            // b*HW + p
  int b = row / HW, p = row % HW;
  float v = x[((size_t)b * C + tid) * HW + p];
  float outv = v;
  if (do_norm) {
    red[tid] = v * v;
    __syncthreads();
    for (int s = 256; s > 0; s >>= 1) {
      if (tid < s) red[tid] += red[tid + s];
      __syncthreads();
    }
    float ms = red[0] / (float)C;
    outv = v * scale[(size_t)b * C + tid] * rsqrtf(ms + 1e-6f);
  }
  dst[(size_t)row * C + tid] = (_Float16)outv;
}

// ---------------------------------------------------------------------------
// Generic f16 WMMA GEMM: C(MxN,f32) = A(MxK,f16,row) * B(KxN,f16,row)
// block tile 64(M) x 128(N), 8 waves in a 4(M) x 2(N) grid; each wave computes
// a 16x64 strip (4 accumulators), reusing one A fragment across 4 WMMAs.
// M%64==0, N%128==0, K%32==0.
// ---------------------------------------------------------------------------
__global__ void __launch_bounds__(256)
k_gemm_f16(const _Float16* __restrict__ A, const _Float16* __restrict__ B,
           float* __restrict__ C, int M, int N, int K) {
  __shared__ _Float16 ldsA[64 * 32];       // [m][k]
  __shared__ _Float16 ldsB[128 * 32];      // transposed: [n][k]
  const int tid  = threadIdx.x;
  const int lane = tid & 31;
  const int wave = tid >> 5;
  const int wm = wave >> 1;                // 0..3: M sub-tile
  const int wn = wave & 1;                 // 0..1: N 64-strip
  const int m0 = blockIdx.y * 64;
  const int n0 = blockIdx.x * 128;
  const bool hi = lane >= 16;
  const int lq = lane & 15;
  v8f acc[4] = {};
  for (int k0 = 0; k0 < K; k0 += 32) {
    // stage A tile: 64x32 halves = 1024 dwords; uint4 per thread (coalesced)
    {
      int row = tid >> 2;                  // 0..63
      int seg = (tid & 3) * 8;             // half offset within row
      *(uint4*)&ldsA[row * 32 + seg] =
          *(const uint4*)&A[(size_t)(m0 + row) * K + k0 + seg];
    }
    // stage B tile transposed: thread covers (k = tid>>3, 16 consecutive n)
    {
      int k  = tid >> 3;
      int ng = (tid & 7) << 4;
      const _Float16* src = &B[(size_t)(k0 + k) * N + n0 + ng];
#pragma unroll
      for (int i = 0; i < 16; ++i) ldsB[(ng + i) * 32 + k] = src[i];
    }
    if (k0 + 32 < K) {
      __builtin_prefetch(&A[(size_t)(m0 + (tid >> 2)) * K + k0 + 32]);
      __builtin_prefetch(&B[(size_t)(k0 + 32 + (tid >> 3)) * N + n0]);
    }
    __syncthreads();
    AF af;
#pragma unroll
    for (int j = 0; j < 8; ++j) {
      int ka = 2 * j + (j >= 4 ? 8 : 0) + (hi ? 8 : 0);
      af.u[j] = *(const unsigned int*)&ldsA[(wm * 16 + lq) * 32 + ka];
    }
#pragma unroll
    for (int t = 0; t < 4; ++t) {
      AF bf;
#pragma unroll
      for (int j = 0; j < 8; ++j) {
        int kb = 2 * j + (j >= 4 ? 8 : 0) + (hi ? 8 : 0);
        bf.u[j] = *(const unsigned int*)&ldsB[(wn * 64 + t * 16 + lq) * 32 + kb];
      }
      acc[t] = wmma16(af, bf, acc[t]);
    }
    __syncthreads();
  }
#pragma unroll
  for (int t = 0; t < 4; ++t) {
    int col = n0 + wn * 64 + t * 16 + lq;
#pragma unroll
    for (int r = 0; r < 8; ++r) {
      int row = m0 + wm * 16 + r + (hi ? 8 : 0);
      C[(size_t)row * N + col] = acc[t][r];
    }
  }
}

// ---------------------------------------------------------------------------
// Cosine-scale + axial RoPE. One block (512 thr) per token row.
// src row layout: [q(512)]? k(512) v(512)  (q present iff has_q)
// writes: q  -> [bh][HW][64]      f16
//         k  -> [bh][NKV][64]     f16 at kv_off + p
//         v  -> [bh][64][NKV] (T) f16 at column kv_off + p
// ---------------------------------------------------------------------------
__global__ void __launch_bounds__(512)
k_qk_rope(const float* __restrict__ src, int row_stride, int has_q,
          const float* __restrict__ pos, const float* __restrict__ scale_arr,
          _Float16* __restrict__ qdst, _Float16* __restrict__ kdst,
          _Float16* __restrict__ vdst, int kv_off, int HW, int NKV, int NH) {
  __shared__ float red[512];
  __shared__ float sq[512];
  __shared__ float sk[512];
  int tid = threadIdx.x;
  int hh = tid >> 6, dd = tid & 63;
  int row = blockIdx.x;            // b*HW + p
  int b = row / HW, p = row % HW;
  const float* r0 = src + (size_t)row * row_stride;
  float qv = has_q ? r0[tid] : 0.0f;
  float kv = r0[(has_q ? 512 : 0) + tid];
  float vv = r0[(has_q ? 1024 : 512) + tid];

  // cosine scale for k
  red[tid] = kv * kv;
  __syncthreads();
  for (int s = 32; s > 0; s >>= 1) {
    if (dd < s) red[tid] += red[tid + s];
    __syncthreads();
  }
  float kss = red[hh * 64];
  __syncthreads();
  float root = sqrtf(scale_arr[hh]);
  float ks_ = kv * root * rsqrtf(kss + 1e-6f);

  float qs_ = 0.0f;
  if (has_q) {
    red[tid] = qv * qv;
    __syncthreads();
    for (int s = 32; s > 0; s >>= 1) {
      if (dd < s) red[tid] += red[tid + s];
      __syncthreads();
    }
    float qss = red[hh * 64];
    __syncthreads();
    qs_ = qv * root * rsqrtf(qss + 1e-6f);
  }

  sq[tid] = qs_;
  sk[tid] = ks_;
  __syncthreads();

  float px = pos[(size_t)row * 2 + 0];
  float py = pos[(size_t)row * 2 + 1];
  float qo = qs_, ko = ks_;
  if (dd < 32) {
    int j = (dd < 16) ? dd : dd - 16;     // theta index 0..15
    int i = (j < 8) ? j : j - 8;          // freq index 0..7
    // f[idx] = exp(log(pi) + idx*log(10)/64), idx = i*8 + head
    float fr = expf(1.14472989f + 0.03597789f * (float)(i * 8 + hh));
    float th = ((j < 8) ? px : py) * fr;
    float c = cosf(th), s = sinf(th);
    int base = hh * 64;
    if (dd < 16) {
      qo = sq[base + dd] * c - sq[base + dd + 16] * s;
      ko = sk[base + dd] * c - sk[base + dd + 16] * s;
    } else {
      qo = sq[base + dd] * c + sq[base + dd - 16] * s;
      ko = sk[base + dd] * c + sk[base + dd - 16] * s;
    }
  }
  int bh = b * NH + hh;
  if (has_q) qdst[((size_t)bh * HW + p) * 64 + dd] = (_Float16)qo;
  kdst[((size_t)bh * NKV + kv_off + p) * 64 + dd] = (_Float16)ko;
  vdst[((size_t)bh * 64 + dd) * NKV + kv_off + p] = (_Float16)vv;
}

// ---------------------------------------------------------------------------
// Flash attention. One wave per (b,h, 16-query tile).
// Scores computed transposed (S^T = K_tile * Q^T) so softmax is per-lane
// (+ one shfl_xor(16) to combine the two half-lane row groups).
// Output accumulated as O^T = V^T * P^T via 4 WMMAs per 32-wide KV tile.
// ---------------------------------------------------------------------------
__global__ void __launch_bounds__(32)
k_attention(const _Float16* __restrict__ Q, const _Float16* __restrict__ Kc,
            const _Float16* __restrict__ Vt, _Float16* __restrict__ O,
            int HW, int NKV, int NH, int INNER) {
  const int lane = threadIdx.x;
  const bool hi = lane >= 16;
  const int lq = lane & 15;            // this lane's query column
  const int bh = blockIdx.y;
  const int b = bh / NH, h = bh % NH;
  const int p0 = blockIdx.x * 16;
  const _Float16* qb = Q + ((size_t)bh * HW + p0) * 64;
  const _Float16* kb = Kc + (size_t)bh * NKV * 64;
  const _Float16* vb = Vt + (size_t)bh * 64 * NKV;

  // Q^T fragments (B operand), two K-steps over d_head=64
  AF qf[2];
#pragma unroll
  for (int ks = 0; ks < 2; ++ks)
#pragma unroll
    for (int j = 0; j < 8; ++j) {
      int d = 2 * j + (j >= 4 ? 8 : 0) + (hi ? 8 : 0) + 32 * ks;
      qf[ks].u[j] = *(const unsigned int*)&qb[(size_t)lq * 64 + d];
    }

  float m_run = -1e30f, l_run = 0.0f;
  v8f oacc[4] = {};                    // O^T, d-chunks of 16

  for (int n0 = 0; n0 < NKV; n0 += 32) {
    v8f s0v = {}, s1v = {};
#pragma unroll
    for (int ks = 0; ks < 2; ++ks) {
      AF a0, a1;
#pragma unroll
      for (int j = 0; j < 8; ++j) {
        int d = 2 * j + (j >= 4 ? 8 : 0) + (hi ? 8 : 0) + 32 * ks;
        a0.u[j] = *(const unsigned int*)&kb[(size_t)(n0 + lq) * 64 + d];
        a1.u[j] = *(const unsigned int*)&kb[(size_t)(n0 + 16 + lq) * 64 + d];
      }
      s0v = wmma16(a0, qf[ks], s0v);   // kv rows n0..n0+15  x q cols
      s1v = wmma16(a1, qf[ks], s1v);   // kv rows n0+16..+31 x q cols
    }
    // online softmax over kv for this lane's query column
    float tmax = -1e30f;
#pragma unroll
    for (int r = 0; r < 8; ++r) tmax = fmaxf(tmax, fmaxf(s0v[r], s1v[r]));
    tmax = fmaxf(tmax, __shfl_xor(tmax, 16));
    float m_new = fmaxf(m_run, tmax);
    float scal = expf(m_run - m_new);
    float P0[8], P1[8], rsum = 0.0f;
#pragma unroll
    for (int r = 0; r < 8; ++r) {
      P0[r] = expf(s0v[r] - m_new);
      P1[r] = expf(s1v[r] - m_new);
      rsum += P0[r] + P1[r];
    }
    rsum += __shfl_xor(rsum, 16);
    l_run = l_run * scal + rsum;
    m_run = m_new;
#pragma unroll
    for (int c4 = 0; c4 < 4; ++c4)
#pragma unroll
      for (int r = 0; r < 8; ++r) oacc[c4][r] *= scal;

    // repack P^T (32 kv x 16 q) into a B fragment in-register
    AF pb;
#pragma unroll
    for (int r = 0; r < 8; ++r) {
      float q0 = __shfl_xor(P0[r], 16);
      float q1 = __shfl_xor(P1[r], 16);
      pb.h[r]     = (_Float16)(hi ? q1 : P0[r]);
      pb.h[8 + r] = (_Float16)(hi ? P1[r] : q0);
    }
    // O^T += V^T(16d x 32kv) * P^T(32kv x 16q), four d-chunks
#pragma unroll
    for (int c4 = 0; c4 < 4; ++c4) {
      AF vf;
#pragma unroll
      for (int j = 0; j < 8; ++j) {
        int kv = 2 * j + (hi ? 16 : 0);
        vf.u[j] = *(const unsigned int*)&vb[(size_t)(c4 * 16 + lq) * NKV + n0 + kv];
      }
      oacc[c4] = wmma16(vf, pb, oacc[c4]);
    }
  }
  float inv = 1.0f / l_run;
#pragma unroll
  for (int c4 = 0; c4 < 4; ++c4)
#pragma unroll
    for (int r = 0; r < 8; ++r) {
      int d = c4 * 16 + r + (hi ? 8 : 0);
      O[((size_t)b * HW + p0 + lq) * INNER + h * 64 + d] =
          (_Float16)(oacc[c4][r] * inv);
    }
}

// ---------------------------------------------------------------------------
// dst(b,C,HW) = x(b,C,HW) + t(b*HW, C)^T
// ---------------------------------------------------------------------------
__global__ void k_residual(const float* __restrict__ x, const float* __restrict__ t,
                           float* __restrict__ dst, int C, int HW, int total) {
  int idx = blockIdx.x * blockDim.x + threadIdx.x;
  if (idx >= total) return;
  int p = idx % HW;
  int bc = idx / HW;
  int ch = bc % C;
  int b = bc / C;
  dst[idx] = x[idx] + t[((size_t)b * HW + p) * C + ch];
}

// ---------------------------------------------------------------------------
// GEGLU: dst[row,i] = a * gelu_exact(g), a=up[row,i], g=up[row,DFF+i]
// ---------------------------------------------------------------------------
__global__ void k_geglu(const float* __restrict__ up, _Float16* __restrict__ dst,
                        int DFF, int total) {
  int idx = blockIdx.x * blockDim.x + threadIdx.x;
  if (idx >= total) return;
  int row = idx / DFF, i = idx % DFF;
  float a = up[(size_t)row * 2 * DFF + i];
  float g = up[(size_t)row * 2 * DFF + DFF + i];
  float gelu = 0.5f * g * (1.0f + erff(g * 0.70710678f));
  dst[idx] = (_Float16)(a * gelu);
}

// ---------------------------------------------------------------------------
// Host orchestration
// ---------------------------------------------------------------------------
extern "C" void kernel_launch(void* const* d_in, const int* in_sizes, int n_in,
                              void* d_out, int out_size, void* d_ws, size_t ws_size,
                              hipStream_t stream) {
  (void)in_sizes; (void)n_in; (void)out_size; (void)ws_size;
  const float* x        = (const float*)d_in[0];
  const float* pos      = (const float*)d_in[1];
  const float* gcond    = (const float*)d_in[2];
  const float* context  = (const float*)d_in[3];
  const float* cpos     = (const float*)d_in[4];
  const float* norm1_w  = (const float*)d_in[5];
  const float* qkv_w    = (const float*)d_in[6];
  const float* x_scale  = (const float*)d_in[7];
  const float* condkv_w = (const float*)d_in[8];
  const float* c_scale  = (const float*)d_in[9];
  const float* proj_w   = (const float*)d_in[10];
  const float* ffn_w    = (const float*)d_in[11];
  const float* ffup_w   = (const float*)d_in[12];
  const float* ffdn_w   = (const float*)d_in[13];
  float* out = (float*)d_out;

  constexpr int B = 2, C = 512, HW = 1024, G = 768, NH = 8;
  constexpr int INNER = 512, NKV = 2 * HW, DFF = 1024, M = B * HW;

  char* base = (char*)d_ws;
  size_t off = 0;
  auto alloc = [&](size_t bytes) -> char* {
    char* p = base + off;
    off += (bytes + 255) & ~(size_t)255;
    return p;
  };
  // persistent f16 weights (~6 MB)
  _Float16* w_qkv  = (_Float16*)alloc((size_t)C * 3 * INNER * 2);
  _Float16* w_kv   = (_Float16*)alloc((size_t)C * 2 * INNER * 2);
  _Float16* w_proj = (_Float16*)alloc((size_t)INNER * C * 2);
  _Float16* w_up   = (_Float16*)alloc((size_t)C * 2 * DFF * 2);
  _Float16* w_dn   = (_Float16*)alloc((size_t)DFF * C * 2);
  float* s1 = (float*)alloc((size_t)B * C * 4);
  float* sf = (float*)alloc((size_t)B * C * 4);
  // tokens (xn region reused later for attention output o_h)
  char* xn_region  = alloc((size_t)M * C * 2);
  _Float16* xn_h   = (_Float16*)xn_region;
  _Float16* o_h    = (_Float16*)xn_region;            // reuse after QKV GEMM
  _Float16* ctx_h  = (_Float16*)alloc((size_t)M * C * 2);
  // big1: qkv_f32 (12MB) then up_f32 (16MB)
  char* big1 = alloc((size_t)M * 2 * DFF * 4);
  float* qkv_f32 = (float*)big1;
  float* up_f32  = (float*)big1;
  // big2: kv_f32 (8MB) then [o_attn (4MB) | down (4MB)]
  char* big2 = alloc((size_t)M * 2 * INNER * 4);
  float* kv_f32  = (float*)big2;
  float* o_attn  = (float*)big2;
  float* dn_f32  = (float*)(big2 + (size_t)M * C * 4);
  // attention operands (reused for FF intermediates after attention)
  char* q_region = alloc((size_t)B * NH * HW * 64 * 2);       // 2 MB
  _Float16* q_h  = (_Float16*)q_region;
  _Float16* xf_h = (_Float16*)q_region;                        // reuse
  char* k_region = alloc((size_t)B * NH * NKV * 64 * 2);      // 4 MB
  _Float16* kall = (_Float16*)k_region;
  _Float16* h_h  = (_Float16*)k_region;                        // reuse
  _Float16* vT   = (_Float16*)alloc((size_t)B * NH * NKV * 64 * 2);
  float* x2 = (float*)alloc((size_t)B * C * HW * 4);

  // 1) weight conversion
  k_convert_w<<<(C * 3 * INNER + 255) / 256, 256, 0, stream>>>(qkv_w, w_qkv, C, 3 * INNER, 0);
  k_convert_w<<<(C * 2 * INNER + 255) / 256, 256, 0, stream>>>(condkv_w, w_kv, C, 2 * INNER, 0);
  k_convert_w<<<(INNER * C + 255) / 256, 256, 0, stream>>>(proj_w, w_proj, INNER, C, 1);
  k_convert_w<<<(C * 2 * DFF + 255) / 256, 256, 0, stream>>>(ffup_w, w_up, C, 2 * DFF, 0);
  k_convert_w<<<(DFF * C + 255) / 256, 256, 0, stream>>>(ffdn_w, w_dn, DFF, C, 0);
  // 2) conditional scales
  k_vecmat_bias1<<<(B * C + 255) / 256, 256, 0, stream>>>(gcond, norm1_w, s1, B, G, C);
  k_vecmat_bias1<<<(B * C + 255) / 256, 256, 0, stream>>>(gcond, ffn_w, sf, B, G, C);
  // 3) tokens: rms-normed x, raw context
  k_to_tokens_f16<<<M, 512, 0, stream>>>(x, s1, xn_h, C, HW, 1);
  k_to_tokens_f16<<<M, 512, 0, stream>>>(context, nullptr, ctx_h, C, HW, 0);
  // 4) QKV and cond-KV GEMMs
  {
    dim3 g(3 * INNER / 128, M / 64);
    k_gemm_f16<<<g, 256, 0, stream>>>(xn_h, w_qkv, qkv_f32, M, 3 * INNER, C);
  }
  {
    dim3 g(2 * INNER / 128, M / 64);
    k_gemm_f16<<<g, 256, 0, stream>>>(ctx_h, w_kv, kv_f32, M, 2 * INNER, C);
  }
  // 5) cosine-scale + RoPE, build attention operands
  k_qk_rope<<<M, 512, 0, stream>>>(qkv_f32, 3 * INNER, 1, pos, x_scale,
                                   q_h, kall, vT, 0, HW, NKV, NH);
  k_qk_rope<<<M, 512, 0, stream>>>(kv_f32, 2 * INNER, 0, cpos, c_scale,
                                   q_h, kall, vT, HW, HW, NKV, NH);
  // 6) flash attention
  {
    dim3 g(HW / 16, B * NH);
    k_attention<<<g, 32, 0, stream>>>(q_h, kall, vT, o_h, HW, NKV, NH, INNER);
  }
  // 7) output projection + residual
  {
    dim3 g(C / 128, M / 64);
    k_gemm_f16<<<g, 256, 0, stream>>>(o_h, w_proj, o_attn, M, C, INNER);
  }
  int tot = B * C * HW;
  k_residual<<<(tot + 255) / 256, 256, 0, stream>>>(x, o_attn, x2, C, HW, tot);
  // 8) FF: norm, up, geglu, down, residual
  k_to_tokens_f16<<<M, 512, 0, stream>>>(x2, sf, xf_h, C, HW, 1);
  {
    dim3 g(2 * DFF / 128, M / 64);
    k_gemm_f16<<<g, 256, 0, stream>>>(xf_h, w_up, up_f32, M, 2 * DFF, C);
  }
  int ng = M * DFF;
  k_geglu<<<(ng + 255) / 256, 256, 0, stream>>>(up_f32, h_h, DFF, ng);
  {
    dim3 g(C / 128, M / 64);
    k_gemm_f16<<<g, 256, 0, stream>>>(h_h, w_dn, dn_f32, M, C, DFF);
  }
  k_residual<<<(tot + 255) / 256, 256, 0, stream>>>(x2, dn_f32, out, C, HW, tot);
}
